// diffRNN_19361712571045
// MI455X (gfx1250) — matrix operand order repaired and checked by reading, not compiled, not run on hardware
//
#include <hip/hip_runtime.h>
#include <hip/hip_bf16.h>
#include <math.h>

// ---------------------------------------------------------------------------
// CDNA5 / gfx1250 bf16 WMMA helpers (wave32)
// ---------------------------------------------------------------------------
typedef __bf16 bf16_t;
typedef __attribute__((ext_vector_type(16))) __bf16 bf16x16;
typedef __attribute__((ext_vector_type(8)))  float  f32x8;

#define LSTR 40  // LDS row stride (bf16 elems) for 32-wide K tiles (bank-spread)
#define F8Z {0.f,0.f,0.f,0.f,0.f,0.f,0.f,0.f}

#define WMMA_BF16(a, b, c) \
  __builtin_amdgcn_wmma_f32_16x16x32_bf16(false, (a), false, (b), (short)0, (c), false, false)

// A fragment: 16x32 (MxK) tile, row-major in LDS with row stride ld.
// Layout: lanes 0-15 -> M=lane, K 0..15; lanes 16-31 -> K 16..31 (matches B frags).
__device__ __forceinline__ bf16x16 frag_a(const bf16_t* As, int ld) {
  const int lane = threadIdx.x & 31;
  const int row  = lane & 15;
  const int koff = (lane & 16) ? 16 : 0;
  const bf16_t* r = As + row * ld;
  bf16x16 a;
#pragma unroll
  for (int j = 0; j < 8; ++j) {
    int k = 2 * j + koff;
    a[2 * j]     = r[k];
    a[2 * j + 1] = r[k + 1];
  }
  return a;
}

// B fragment from [N,K]-row-major tile (i.e. W stored as [out,in], y = x @ W^T).
// Layout used here (contiguous k, same as frag_a): lane -> column n = lane&15 ; comps c -> K = c + 16*(lane>=16).
__device__ __forceinline__ bf16x16 frag_b_nk(const bf16_t* Bs, int ld) {
  const int lane = threadIdx.x & 31;
  const int n  = lane & 15;
  const int kb = (lane & 16) ? 16 : 0;
  const bf16_t* r = Bs + n * ld + kb;
  bf16x16 b;
#pragma unroll
  for (int c = 0; c < 16; ++c) b[c] = r[c];
  return b;
}

// B fragment from [K,N]-row-major tile.
__device__ __forceinline__ bf16x16 frag_b_kn(const bf16_t* Bs, int ld) {
  const int lane = threadIdx.x & 31;
  const int n  = lane & 15;
  const int kb = (lane & 16) ? 16 : 0;
  bf16x16 b;
#pragma unroll
  for (int c = 0; c < 16; ++c) b[c] = Bs[(kb + c) * ld + n];
  return b;
}

// ---------------------------------------------------------------------------
// Elementwise / small kernels
// ---------------------------------------------------------------------------
__global__ void k_f32_to_bf16(const float* __restrict__ src, bf16_t* __restrict__ dst, int n) {
  int i = blockIdx.x * blockDim.x + threadIdx.x;
  if (i < n) dst[i] = (bf16_t)src[i];
}

// ve = relu(emb_table[input_seqs])  -> bf16 [B,V,S,D]
__global__ void k_embed(const int* __restrict__ seqs, const float* __restrict__ emb,
                        bf16_t* __restrict__ ve, int total) {
  int i = blockIdx.x * blockDim.x + threadIdx.x;
  if (i >= total) return;
  int tok = seqs[i >> 8];
  float v = emb[tok * 256 + (i & 255)];
  ve[i] = (bf16_t)fmaxf(v, 0.f);
}

// temb[b][o] = timestep_embedding(t,128) @ temb_w^T + temb_b
__global__ void k_temb(const int* __restrict__ tstep, const float* __restrict__ w,
                       const float* __restrict__ bvec, float* __restrict__ temb) {
  __shared__ float e[128];
  const int b = blockIdx.x;
  const int t = threadIdx.x;   // 0..127 = output index
  if (t < 64) {
    float fr = __expf(-logf(10000.f) * (float)t / 64.f);
    float a  = (float)tstep[b] * fr;
    e[t]      = __sinf(a);
    e[64 + t] = __cosf(a);
  }
  __syncthreads();
  float s = bvec[t];
  for (int k = 0; k < 128; ++k) s += e[k] * w[t * 128 + k];
  temb[b * 128 + t] = s;
}

// ---------------------------------------------------------------------------
// Conv1 implicit-GEMM: in [B,50,20,256] -> h1 [B,128,20,256] bf16
// fused  + conv1_b + temb broadcast + SiLU
// ---------------------------------------------------------------------------
__global__ __launch_bounds__(256) void k_conv1(
    const bf16_t* __restrict__ ve, const bf16_t* __restrict__ w,
    const float* __restrict__ bias, const float* __restrict__ temb,
    bf16_t* __restrict__ h1) {
  __shared__ bf16_t As[32 * LSTR];
  __shared__ bf16_t Bs[64 * LSTR];
  const int t  = threadIdx.x;
  const int m0 = blockIdx.x * 32;  // m = (b,y,x) over 327680
  const int n0 = blockIdx.y * 64;  // co over 128
  const int wave = t >> 5, wr = wave >> 2, wc = wave & 3;
  f32x8 acc = F8Z;
  for (int ch = 0; ch < 18; ++ch) {  // 9 taps x (50->64 ci, 2 chunks of 32)
    const int tap = ch >> 1;
    const int ci0 = (ch & 1) << 5;
    const int dy = tap / 3 - 1, dx = tap % 3 - 1;
    for (int i = t; i < 32 * 32; i += 256) {
      int ml = i & 31, kk = i >> 5;
      int ci = ci0 + kk;
      int m = m0 + ml;
      int b = m / 5120, rem = m % 5120;
      int y = rem >> 8, x = rem & 255;
      int yy = y + dy, xx = x + dx;
      bf16_t v = (bf16_t)0.f;
      if (ci < 50 && yy >= 0 && yy < 20 && xx >= 0 && xx < 256)
        v = ve[((b * 50 + ci) * 20 + yy) * 256 + xx];
      As[ml * LSTR + kk] = v;
    }
    for (int i = t; i < 64 * 32; i += 256) {
      int n = i >> 5, kk = i & 31;
      int ci = ci0 + kk;
      bf16_t v = (bf16_t)0.f;
      if (ci < 50) v = w[((n0 + n) * 50 + ci) * 9 + tap];
      Bs[n * LSTR + kk] = v;
    }
    __syncthreads();
    bf16x16 a  = frag_a(As + wr * 16 * LSTR, LSTR);
    bf16x16 bb = frag_b_nk(Bs + wc * 16 * LSTR, LSTR);
    acc = WMMA_BF16(a, bb, acc);
    __syncthreads();
  }
  const int lane = t & 31;
  const int col = lane & 15;
  const int rb  = (lane & 16) ? 8 : 0;
  const int co  = n0 + wc * 16 + col;
#pragma unroll
  for (int r = 0; r < 8; ++r) {
    int m = m0 + wr * 16 + rb + r;
    int b = m / 5120, rem = m % 5120;
    int y = rem >> 8, x = rem & 255;
    float v = acc[r] + bias[co] + temb[b * 128 + co];
    v = v / (1.f + __expf(-v));   // SiLU
    h1[((b * 128 + co) * 20 + y) * 256 + x] = (bf16_t)v;
  }
}

// ---------------------------------------------------------------------------
// Conv2 implicit-GEMM: h1 [B,128,20,256] -> predicted_noise fp32 (d_out slot)
// fused epilogue: gen1 = relu(emb[seq]) + noise - pred  (bf16)
// ---------------------------------------------------------------------------
__global__ __launch_bounds__(256) void k_conv2(
    const bf16_t* __restrict__ h1, const bf16_t* __restrict__ w,
    const float* __restrict__ bias, const int* __restrict__ seqs,
    const float* __restrict__ emb, const float* __restrict__ noise,
    float* __restrict__ pred, bf16_t* __restrict__ gen1) {
  __shared__ bf16_t As[32 * LSTR];
  __shared__ bf16_t Bs[64 * LSTR];
  const int t  = threadIdx.x;
  const int m0 = blockIdx.x * 32;
  const int wave = t >> 5, wr = wave >> 2, wc = wave & 3;
  f32x8 acc = F8Z;
  for (int ch = 0; ch < 36; ++ch) {  // 9 taps x (128 ci / 32)
    const int tap = ch >> 2;
    const int ci0 = (ch & 3) << 5;
    const int dy = tap / 3 - 1, dx = tap % 3 - 1;
    for (int i = t; i < 32 * 32; i += 256) {
      int ml = i & 31, kk = i >> 5;
      int ci = ci0 + kk;
      int m = m0 + ml;
      int b = m / 5120, rem = m % 5120;
      int y = rem >> 8, x = rem & 255;
      int yy = y + dy, xx = x + dx;
      bf16_t v = (bf16_t)0.f;
      if (yy >= 0 && yy < 20 && xx >= 0 && xx < 256)
        v = h1[((b * 128 + ci) * 20 + yy) * 256 + xx];
      As[ml * LSTR + kk] = v;
    }
    for (int i = t; i < 64 * 32; i += 256) {
      int n = i >> 5, kk = i & 31;
      int ci = ci0 + kk;
      bf16_t v = (bf16_t)0.f;
      if (n < 50) v = w[(n * 128 + ci) * 9 + tap];
      Bs[n * LSTR + kk] = v;
    }
    __syncthreads();
    bf16x16 a  = frag_a(As + wr * 16 * LSTR, LSTR);
    bf16x16 bb = frag_b_nk(Bs + wc * 16 * LSTR, LSTR);
    acc = WMMA_BF16(a, bb, acc);
    __syncthreads();
  }
  const int lane = t & 31;
  const int col = lane & 15;
  const int rb  = (lane & 16) ? 8 : 0;
  const int co  = wc * 16 + col;   // 0..63, valid < 50
#pragma unroll
  for (int r = 0; r < 8; ++r) {
    if (co < 50) {
      int m = m0 + wr * 16 + rb + r;
      int b = m / 5120, rem = m % 5120;
      int y = rem >> 8, x = rem & 255;
      int idx = ((b * 50 + co) * 20 + y) * 256 + x;
      float p = acc[r] + bias[co];
      pred[idx] = p;
      int tok = seqs[(b * 50 + co) * 20 + y];
      float v = fmaxf(emb[tok * 256 + x], 0.f);
      gen1[idx] = (bf16_t)(v + noise[idx] - p);
    }
  }
}

// ---------------------------------------------------------------------------
// Attention: fused QKV projection (per visit). q scaled by 1/sqrt(32).
// outputs qb/kb/vb bf16 laid out [(n*8+h)][l=64][dh=32]
// ---------------------------------------------------------------------------
__global__ __launch_bounds__(256) void k_qkv(
    const bf16_t* __restrict__ xq, int qvis, int qzero,
    const bf16_t* __restrict__ xkv, int kvis,
    const bf16_t* __restrict__ w, const float* __restrict__ bias,
    bf16_t* __restrict__ qb, bf16_t* __restrict__ kb, bf16_t* __restrict__ vb) {
  __shared__ bf16_t As[32 * LSTR];
  __shared__ bf16_t Bs[64 * LSTR];
  const int t  = threadIdx.x;
  const int m0 = blockIdx.x * 32;   // m = l*20+n over 1280
  const int n0 = blockIdx.y * 64;   // o over 768
  const int kind = n0 >> 8;         // 0=q 1=k 2=v (64-tile never crosses 256)
  const bf16_t* xb = (kind == 0) ? xq : xkv;
  const int vis  = (kind == 0) ? qvis : kvis;
  const int zero = (kind == 0) ? qzero : 0;
  const int wave = t >> 5, wr = wave >> 2, wc = wave & 3;
  f32x8 acc = F8Z;
  for (int k0 = 0; k0 < 256; k0 += 32) {
    for (int i = t; i < 32 * 32; i += 256) {
      int ml = i & 31, kk = i >> 5;
      int m = m0 + ml;
      bf16_t v = (bf16_t)0.f;
      if (!zero) {
        int l = m / 20, n = m % 20;
        v = xb[((l * 50 + vis) * 20 + n) * 256 + k0 + kk];
      }
      As[ml * LSTR + kk] = v;
    }
    for (int i = t; i < 64 * 32; i += 256) {
      int n = i >> 5, kk = i & 31;
      Bs[n * LSTR + kk] = w[(n0 + n) * 256 + k0 + kk];
    }
    __syncthreads();
    bf16x16 a  = frag_a(As + wr * 16 * LSTR, LSTR);
    bf16x16 bb = frag_b_nk(Bs + wc * 16 * LSTR, LSTR);
    acc = WMMA_BF16(a, bb, acc);
    __syncthreads();
  }
  const int lane = t & 31;
  const int col = lane & 15;
  const int rb  = (lane & 16) ? 8 : 0;
  const int o   = n0 + wc * 16 + col;
  const int og  = o & 255;
  const int h = og >> 5, dh = og & 31;
  bf16_t* dst = (kind == 0) ? qb : (kind == 1) ? kb : vb;
  const float scale = (kind == 0) ? 0.17677669529663687f : 1.f;
#pragma unroll
  for (int r = 0; r < 8; ++r) {
    int m = m0 + wr * 16 + rb + r;
    int l = m / 20, n = m % 20;
    float v = (acc[r] + bias[o]) * scale;
    dst[((n * 8 + h) * 64 + l) * 32 + dh] = (bf16_t)v;
  }
}

// scores (WMMA) -> softmax over keys -> P@V (WMMA). One block per (n,h).
__global__ __launch_bounds__(256) void k_attn(
    const bf16_t* __restrict__ qb, const bf16_t* __restrict__ kb,
    const bf16_t* __restrict__ vb, bf16_t* __restrict__ ao) {
  __shared__ bf16_t Qs[64 * LSTR];
  __shared__ bf16_t Ks[64 * LSTR];
  __shared__ bf16_t Vs[64 * LSTR];
  __shared__ float  Ss[64 * 68];
  __shared__ bf16_t Ps[64 * 72];
  const int t  = threadIdx.x;
  const int nh = blockIdx.x;           // n*8+h
  const int base = nh * 64 * 32;
  for (int i = t; i < 2048; i += 256) {
    int l = i >> 5, dh = i & 31;
    Qs[l * LSTR + dh] = qb[base + i];
    Ks[l * LSTR + dh] = kb[base + i];
    Vs[l * LSTR + dh] = vb[base + i];
  }
  __syncthreads();
  const int wave = t >> 5;
  const int lane = t & 31;
  const int ccol = lane & 15;
  const int rb   = (lane & 16) ? 8 : 0;
  // S[64][64] = Q @ K^T : 16 tiles, 2 per wave, K=32 single WMMA each
#pragma unroll
  for (int u = 0; u < 2; ++u) {
    int ti = wave * 2 + u;
    int tr = ti >> 2, tc = ti & 3;
    f32x8 acc = F8Z;
    bf16x16 a  = frag_a(Qs + tr * 16 * LSTR, LSTR);
    bf16x16 bb = frag_b_nk(Ks + tc * 16 * LSTR, LSTR);  // B[k=dh][m] = K[m][dh]
    acc = WMMA_BF16(a, bb, acc);
#pragma unroll
    for (int r = 0; r < 8; ++r)
      Ss[(tr * 16 + rb + r) * 68 + tc * 16 + ccol] = acc[r];
  }
  __syncthreads();
  if (t < 64) {  // softmax over m (64) per query row
    float mx = -1e30f;
    for (int m = 0; m < 64; ++m) mx = fmaxf(mx, Ss[t * 68 + m]);
    float sum = 0.f;
    for (int m = 0; m < 64; ++m) {
      float e = __expf(Ss[t * 68 + m] - mx);
      Ss[t * 68 + m] = e;
      sum += e;
    }
    float inv = 1.f / sum;
    for (int m = 0; m < 64; ++m) Ps[t * 72 + m] = (bf16_t)(Ss[t * 68 + m] * inv);
  }
  __syncthreads();
  // O[64][32] = P @ V : 8 tiles (4x2), one per wave, K=64 (2 chunks)
  {
    int tr = wave >> 1, tc = wave & 1;
    f32x8 acc = F8Z;
#pragma unroll
    for (int kc = 0; kc < 2; ++kc) {
      bf16x16 a  = frag_a(Ps + tr * 16 * 72 + kc * 32, 72);
      bf16x16 bb = frag_b_kn(Vs + kc * 32 * LSTR + tc * 16, LSTR);
      acc = WMMA_BF16(a, bb, acc);
    }
    int n_seq = nh >> 3, h = nh & 7;
#pragma unroll
    for (int r = 0; r < 8; ++r) {
      int l  = tr * 16 + rb + r;
      int dh = tc * 16 + ccol;
      ao[(l * 20 + n_seq) * 256 + h * 32 + dh] = (bf16_t)acc[r];
    }
  }
}

// output projection, scattered into gen2 [B,V,S,D] for this visit
__global__ __launch_bounds__(256) void k_outproj(
    const bf16_t* __restrict__ ao, const bf16_t* __restrict__ w,
    const float* __restrict__ bias, int vis, bf16_t* __restrict__ gen2) {
  __shared__ bf16_t As[32 * LSTR];
  __shared__ bf16_t Bs[64 * LSTR];
  const int t  = threadIdx.x;
  const int m0 = blockIdx.x * 32;
  const int n0 = blockIdx.y * 64;
  const int wave = t >> 5, wr = wave >> 2, wc = wave & 3;
  f32x8 acc = F8Z;
  for (int k0 = 0; k0 < 256; k0 += 32) {
    for (int i = t; i < 32 * 32; i += 256) {
      int ml = i & 31, kk = i >> 5;
      As[ml * LSTR + kk] = ao[(m0 + ml) * 256 + k0 + kk];
    }
    for (int i = t; i < 64 * 32; i += 256) {
      int n = i >> 5, kk = i & 31;
      Bs[n * LSTR + kk] = w[(n0 + n) * 256 + k0 + kk];
    }
    __syncthreads();
    bf16x16 a  = frag_a(As + wr * 16 * LSTR, LSTR);
    bf16x16 bb = frag_b_nk(Bs + wc * 16 * LSTR, LSTR);
    acc = WMMA_BF16(a, bb, acc);
    __syncthreads();
  }
  const int lane = t & 31;
  const int col = lane & 15;
  const int rb  = (lane & 16) ? 8 : 0;
  const int o   = n0 + wc * 16 + col;
#pragma unroll
  for (int r = 0; r < 8; ++r) {
    int m = m0 + wr * 16 + rb + r;
    int l = m / 20, n = m % 20;
    gen2[((l * 50 + vis) * 20 + n) * 256 + o] = (bf16_t)(acc[r] + bias[o]);
  }
}

// ---------------------------------------------------------------------------
// LSTM: per-step gates = X_t @ w_ih^T + h @ w_hh^T + b_ih + b_hh  (fp32)
// ---------------------------------------------------------------------------
__global__ __launch_bounds__(256) void k_lstm_gemm(
    const bf16_t* __restrict__ X, int tstep, const bf16_t* __restrict__ hprev,
    const bf16_t* __restrict__ wih, const bf16_t* __restrict__ whh,
    const float* __restrict__ bih, const float* __restrict__ bhh,
    float* __restrict__ gates) {
  __shared__ bf16_t As[32 * LSTR];
  __shared__ bf16_t Bs[64 * LSTR];
  const int t  = threadIdx.x;
  const int m0 = blockIdx.x * 32;   // m over 3200
  const int n0 = blockIdx.y * 64;   // gate col over 1024
  const int wave = t >> 5, wr = wave >> 2, wc = wave & 3;
  f32x8 acc = F8Z;
  for (int phase = 0; phase < 2; ++phase) {
    const bf16_t* wsrc = phase ? whh : wih;
    for (int k0 = 0; k0 < 256; k0 += 32) {
      for (int i = t; i < 32 * 32; i += 256) {
        int ml = i & 31, kk = i >> 5;
        int m = m0 + ml;
        bf16_t v;
        if (phase == 0) v = X[(m * 20 + tstep) * 256 + k0 + kk];
        else            v = hprev[m * 256 + k0 + kk];
        As[ml * LSTR + kk] = v;
      }
      for (int i = t; i < 64 * 32; i += 256) {
        int n = i >> 5, kk = i & 31;
        Bs[n * LSTR + kk] = wsrc[(n0 + n) * 256 + k0 + kk];
      }
      if (k0 + 32 < 256)
        __builtin_prefetch(&wsrc[(n0 + (t >> 3)) * 256 + k0 + 32], 0, 1);
      __syncthreads();
      bf16x16 a  = frag_a(As + wr * 16 * LSTR, LSTR);
      bf16x16 bb = frag_b_nk(Bs + wc * 16 * LSTR, LSTR);
      acc = WMMA_BF16(a, bb, acc);
      __syncthreads();
    }
  }
  const int lane = t & 31;
  const int col = lane & 15;
  const int rb  = (lane & 16) ? 8 : 0;
  const int o   = n0 + wc * 16 + col;
#pragma unroll
  for (int r = 0; r < 8; ++r) {
    int m = m0 + wr * 16 + rb + r;
    gates[m * 1024 + o] = acc[r] + bih[o] + bhh[o];
  }
}

__global__ void k_lstm_pw(const float* __restrict__ gates, float* __restrict__ c,
                          bf16_t* __restrict__ h) {
  int i = blockIdx.x * blockDim.x + threadIdx.x;
  if (i >= 3200 * 256) return;
  int m = i >> 8, hh = i & 255;
  const float* g = gates + m * 1024;
  float gi = g[hh], gf = g[256 + hh], gg = g[512 + hh], go = g[768 + hh];
  float si = 1.f / (1.f + __expf(-gi));
  float sf = 1.f / (1.f + __expf(-gf));
  float so = 1.f / (1.f + __expf(-go));
  float cn = sf * c[i] + si * tanhf(gg);
  c[i] = cn;
  h[i] = (bf16_t)(so * tanhf(cn));
}

__global__ void k_zero_hc(float* __restrict__ c, bf16_t* __restrict__ h, int n) {
  int i = blockIdx.x * blockDim.x + threadIdx.x;
  if (i < n) { c[i] = 0.f; h[i] = (bf16_t)0.f; }
}

// ---------------------------------------------------------------------------
// Classifier GEMM with ReLU: Y[M,N] = relu(X[M,K] @ W[N,K]^T + b), bf16 out
// ---------------------------------------------------------------------------
__global__ __launch_bounds__(256) void k_gemm_relu(
    const bf16_t* __restrict__ X, const bf16_t* __restrict__ W,
    const float* __restrict__ bias, bf16_t* __restrict__ Y, int K, int N) {
  __shared__ bf16_t As[32 * LSTR];
  __shared__ bf16_t Bs[64 * LSTR];
  const int t  = threadIdx.x;
  const int m0 = blockIdx.x * 32;
  const int n0 = blockIdx.y * 64;
  const int wave = t >> 5, wr = wave >> 2, wc = wave & 3;
  f32x8 acc = F8Z;
  for (int k0 = 0; k0 < K; k0 += 32) {
    for (int i = t; i < 32 * 32; i += 256) {
      int ml = i & 31, kk = i >> 5;
      As[ml * LSTR + kk] = X[(m0 + ml) * K + k0 + kk];
    }
    for (int i = t; i < 64 * 32; i += 256) {
      int n = i >> 5, kk = i & 31;
      Bs[n * LSTR + kk] = W[(n0 + n) * K + k0 + kk];
    }
    if (k0 + 32 < K)
      __builtin_prefetch(&X[(m0 + (t >> 3)) * K + k0 + 32], 0, 1);
    __syncthreads();
    bf16x16 a  = frag_a(As + wr * 16 * LSTR, LSTR);
    bf16x16 bb = frag_b_nk(Bs + wc * 16 * LSTR, LSTR);
    acc = WMMA_BF16(a, bb, acc);
    __syncthreads();
  }
  const int lane = t & 31;
  const int col = lane & 15;
  const int rb  = (lane & 16) ? 8 : 0;
  const int o   = n0 + wc * 16 + col;
#pragma unroll
  for (int r = 0; r < 8; ++r) {
    int m = m0 + wr * 16 + rb + r;
    Y[m * N + o] = (bf16_t)fmaxf(acc[r] + bias[o], 0.f);
  }
}

// final 2-way head + softmax; also extracts [:, -1, :]
__global__ void k_cls3(const bf16_t* __restrict__ z2, const float* __restrict__ w3,
                       const float* __restrict__ b3, float* __restrict__ res,
                       float* __restrict__ last) {
  int m = blockIdx.x * blockDim.x + threadIdx.x;
  if (m >= 3200) return;
  float l0 = b3[0], l1 = b3[1];
  for (int k = 0; k < 1024; ++k) {
    float z = (float)z2[m * 1024 + k];
    l0 += z * w3[k];
    l1 += z * w3[1024 + k];
  }
  float mx = fmaxf(l0, l1);
  float e0 = __expf(l0 - mx), e1 = __expf(l1 - mx);
  float inv = 1.f / (e0 + e1);
  float p0 = e0 * inv, p1 = e1 * inv;
  res[m * 2] = p0;
  res[m * 2 + 1] = p1;
  if ((m % 50) == 49) {
    int b = m / 50;
    last[b * 2] = p0;
    last[b * 2 + 1] = p1;
  }
}

// ---------------------------------------------------------------------------
// Orchestration
// ---------------------------------------------------------------------------
extern "C" void kernel_launch(void* const* d_in, const int* in_sizes, int n_in,
                              void* d_out, int out_size, void* d_ws, size_t ws_size,
                              hipStream_t stream) {
  (void)in_sizes; (void)n_in; (void)out_size; (void)ws_size;
  const int*   seqs       = (const int*)d_in[0];
  /* d_in[1] seq_time_step: unused by reference */
  const int*   tstep      = (const int*)d_in[2];
  const float* noise      = (const float*)d_in[3];
  const float* emb        = (const float*)d_in[4];
  const float* attn_in_w  = (const float*)d_in[5];
  const float* attn_in_b  = (const float*)d_in[6];
  const float* attn_out_w = (const float*)d_in[7];
  const float* attn_out_b = (const float*)d_in[8];
  const float* lstm_w_ih  = (const float*)d_in[9];
  const float* lstm_w_hh  = (const float*)d_in[10];
  const float* lstm_b_ih  = (const float*)d_in[11];
  const float* lstm_b_hh  = (const float*)d_in[12];
  const float* cls_w1     = (const float*)d_in[13];
  const float* cls_b1     = (const float*)d_in[14];
  const float* cls_w2     = (const float*)d_in[15];
  const float* cls_b2     = (const float*)d_in[16];
  const float* cls_w3     = (const float*)d_in[17];
  const float* cls_b3     = (const float*)d_in[18];
  const float* temb_w     = (const float*)d_in[19];
  const float* temb_b     = (const float*)d_in[20];
  const float* conv1_w    = (const float*)d_in[21];
  const float* conv1_b    = (const float*)d_in[22];
  const float* conv2_w    = (const float*)d_in[23];
  const float* conv2_b    = (const float*)d_in[24];

  float* out = (float*)d_out;
  // output slots (floats): res | res_gen | last | last_gen | noise | pred
  float* o_res   = out;
  float* o_resg  = out + 6400;
  float* o_last  = out + 12800;
  float* o_lastg = out + 12928;
  float* o_noise = out + 13056;
  float* o_pred  = out + 16397056;

  // workspace carve (256B aligned)
  char* p = (char*)d_ws;
  auto alloc = [&](size_t bytes) -> void* {
    void* r = (void*)p;
    p += (bytes + 255) & ~(size_t)255;
    return r;
  };
  bf16_t* ve_bf   = (bf16_t*)alloc((size_t)16384000 * 2);
  bf16_t* h1_bf   = (bf16_t*)alloc((size_t)41943040 * 2);
  bf16_t* gen1_bf = (bf16_t*)alloc((size_t)16384000 * 2);
  bf16_t* gen2_bf = (bf16_t*)alloc((size_t)16384000 * 2);
  bf16_t* w_in_bf  = (bf16_t*)alloc((size_t)196608 * 2);
  bf16_t* w_out_bf = (bf16_t*)alloc((size_t)65536 * 2);
  bf16_t* w_ih_bf  = (bf16_t*)alloc((size_t)262144 * 2);
  bf16_t* w_hh_bf  = (bf16_t*)alloc((size_t)262144 * 2);
  bf16_t* w_c1_bf  = (bf16_t*)alloc((size_t)262144 * 2);
  bf16_t* w_c2_bf  = (bf16_t*)alloc((size_t)1048576 * 2);
  bf16_t* w_cv1_bf = (bf16_t*)alloc((size_t)57600 * 2);
  bf16_t* w_cv2_bf = (bf16_t*)alloc((size_t)57600 * 2);
  float*  temb_f   = (float*)alloc((size_t)8192 * 4);
  bf16_t* qb = (bf16_t*)alloc((size_t)327680 * 2);
  bf16_t* kb = (bf16_t*)alloc((size_t)327680 * 2);
  bf16_t* vb = (bf16_t*)alloc((size_t)327680 * 2);
  bf16_t* ao = (bf16_t*)alloc((size_t)327680 * 2);
  float*  gates = (float*)alloc((size_t)3276800 * 4);
  bf16_t* h_bf  = (bf16_t*)alloc((size_t)819200 * 2);
  float*  c_f32 = (float*)alloc((size_t)819200 * 4);
  bf16_t* z1_bf = (bf16_t*)alloc((size_t)3276800 * 2);
  bf16_t* z2_bf = (bf16_t*)alloc((size_t)3276800 * 2);

  auto cvt = [&](const float* s, bf16_t* d, int n) {
    k_f32_to_bf16<<<(n + 255) / 256, 256, 0, stream>>>(s, d, n);
  };
  cvt(attn_in_w, w_in_bf, 196608);
  cvt(attn_out_w, w_out_bf, 65536);
  cvt(lstm_w_ih, w_ih_bf, 262144);
  cvt(lstm_w_hh, w_hh_bf, 262144);
  cvt(cls_w1, w_c1_bf, 262144);
  cvt(cls_w2, w_c2_bf, 1048576);
  cvt(conv1_w, w_cv1_bf, 57600);
  cvt(conv2_w, w_cv2_bf, 57600);

  // embedding + relu (bf16), timestep embedding MLP
  k_embed<<<64000, 256, 0, stream>>>(seqs, emb, ve_bf, 16384000);
  k_temb<<<64, 128, 0, stream>>>(tstep, temb_w, temb_b, temb_f);

  // diffusion denoiser (two WMMA implicit-GEMM convs, fused epilogues)
  k_conv1<<<dim3(10240, 2), 256, 0, stream>>>(ve_bf, w_cv1_bf, conv1_b, temb_f, h1_bf);
  k_conv2<<<dim3(10240, 1), 256, 0, stream>>>(h1_bf, w_cv2_bf, conv2_b, seqs, emb,
                                              noise, o_pred, gen1_bf);

  // sequential cross-attention over visits
  for (int vis = 0; vis < 50; ++vis) {
    k_qkv<<<dim3(40, 12), 256, 0, stream>>>(gen2_bf, vis - 1, (vis == 0) ? 1 : 0,
                                            gen1_bf, vis, w_in_bf, attn_in_b,
                                            qb, kb, vb);
    k_attn<<<160, 256, 0, stream>>>(qb, kb, vb, ao);
    k_outproj<<<dim3(40, 4), 256, 0, stream>>>(ao, w_out_bf, attn_out_b, vis, gen2_bf);
  }

  // LSTM (zero-init every visit per reference) + classifier, twice
  auto lstm_cls = [&](const bf16_t* X, float* res, float* last) {
    k_zero_hc<<<3200, 256, 0, stream>>>(c_f32, h_bf, 819200);
    for (int tt = 0; tt < 20; ++tt) {
      k_lstm_gemm<<<dim3(100, 16), 256, 0, stream>>>(X, tt, h_bf, w_ih_bf, w_hh_bf,
                                                     lstm_b_ih, lstm_b_hh, gates);
      k_lstm_pw<<<3200, 256, 0, stream>>>(gates, c_f32, h_bf);
    }
    k_gemm_relu<<<dim3(100, 16), 256, 0, stream>>>(h_bf, w_c1_bf, cls_b1, z1_bf, 256, 1024);
    k_gemm_relu<<<dim3(100, 16), 256, 0, stream>>>(z1_bf, w_c2_bf, cls_b2, z2_bf, 1024, 1024);
    k_cls3<<<13, 256, 0, stream>>>(z2_bf, cls_w3, cls_b3, res, last);
  };
  lstm_cls(ve_bf, o_res, o_last);
  lstm_cls(gen2_bf, o_resg, o_lastg);

  // pass-through output: normal_noise
  hipMemcpyAsync(o_noise, noise, (size_t)16384000 * sizeof(float),
                 hipMemcpyDeviceToDevice, stream);
}
